// MultiLayerBase_1675037245917
// MI455X (gfx1250) — compile-verified, matrix operand order checked by reading, not run on hardware
//
#include <hip/hip_runtime.h>
#include <hip/hip_bf16.h>
#include <math.h>
#include <stdint.h>

// Problem dims (match reference)
#define Bb 32
#define Ss 2048
#define Dd 256
#define Hh 256
#define Gg 768   // 3*H

typedef __attribute__((ext_vector_type(16))) __bf16 v16bf;
typedef __attribute__((ext_vector_type(8)))  __bf16 v8bf;
typedef __attribute__((ext_vector_type(8)))  float  v8f;
typedef __attribute__((ext_vector_type(4)))  unsigned int v4u;
typedef __attribute__((ext_vector_type(8)))  int  v8i;
typedef __attribute__((ext_vector_type(4)))  int  v4i;

// Global-address-space (addrspace 1) view of a bf16x8 fragment: loads through
// this type lower to global_load_b128 (LOADcnt only), not flat_load (which
// also ties up DScnt per ISA 10.1.1).
typedef v8bf __attribute__((address_space(1))) gv8bf;

static __device__ inline float bf2f(__bf16 x) {
    unsigned short s = __builtin_bit_cast(unsigned short, x);
    unsigned u = ((unsigned)s) << 16;
    return __builtin_bit_cast(float, u);
}
static __device__ inline __bf16 f2bf(float f) {
    unsigned u = __builtin_bit_cast(unsigned, f);
    unsigned r = u + 0x7FFFu + ((u >> 16) & 1u);   // round-to-nearest-even
    unsigned short s = (unsigned short)(r >> 16);
    return __builtin_bit_cast(__bf16, s);
}

// ---- CDNA5 async global->LDS copy (ASYNCcnt-tracked, bypasses VGPRs) ------
static __device__ inline void async_copy_b128(uint32_t lds_byte_addr, const void* gptr) {
    asm volatile("global_load_async_to_lds_b128 %0, %1, off"
                 :: "v"(lds_byte_addr), "v"((uint64_t)(uintptr_t)gptr)
                 : "memory");
}
static __device__ inline void wait_async0() {
    asm volatile("s_wait_asynccnt 0x0" ::: "memory");
}

// ---- CDNA5 Tensor Data Mover: 1-D bulk copy global -> LDS -----------------
// D# per ISA ch.8: group0 = {count=1, lds_addr, global_addr[56:0], type=2},
// group1 = {data_size=8B, tensor_dim0 = tile_dim0 = nElem8, tensor_dim1 = 1,
// tensor_dim0_stride = nElem8}; groups 2/3 zero (<=2-D descriptor form).
// One wave issues it; completion tracked with TENSORcnt.
static __device__ inline void tdm_load_1d(uint32_t lds_addr, uint64_t gaddr, uint32_t nElem8) {
    v4u g0;
    g0.x = 1u;                                                   // count=1 (valid user D#)
    g0.y = lds_addr;                                             // LDS byte address
    g0.z = (uint32_t)gaddr;                                      // global_addr[31:0]
    g0.w = (uint32_t)((gaddr >> 32) & 0x01FFFFFFu) | 0x80000000u; // [56:32] | type=2
    v8i g1;
    g1[0] = (int)(3u << 16);                                     // data_size = 8 bytes
    g1[1] = (int)(nElem8 << 16);                                 // tensor_dim0[15:0]
    g1[2] = (int)(((nElem8 >> 16) & 0xFFFFu) | (1u << 16));      // dim0 hi | tensor_dim1=1
    g1[3] = (int)(nElem8 << 16);                                 // dim1 hi=0 | tile_dim0
    g1[4] = 1;                                                   // tile_dim1=1 | tile_dim2=0
    g1[5] = (int)nElem8;                                         // tensor_dim0_stride lo
    g1[6] = 0;                                                   // stride hi | dim1_stride lo
    g1[7] = 0;                                                   // dim1_stride hi
    v4i gz4 = {0, 0, 0, 0};
    v8i gz8 = {0, 0, 0, 0, 0, 0, 0, 0};
    __builtin_amdgcn_tensor_load_to_lds(g0, g1, gz4, gz4, gz8, 0);
}

// ---------------------------------------------------------------------------
// fp32 -> bf16 conversion (one-time, bandwidth-trivial)
// ---------------------------------------------------------------------------
__global__ void cvt_f32_bf16(const float* __restrict__ in, __bf16* __restrict__ out, size_t n) {
    size_t i = (size_t)blockIdx.x * blockDim.x + threadIdx.x;
    size_t stride = (size_t)gridDim.x * blockDim.x;
    for (; i < n; i += stride) out[i] = f2bf(in[i]);
}

// ---------------------------------------------------------------------------
// Input-projection GEMM:  gx[tok][g] = sum_k A[tok][k] * W[g][k] + bias[g]
// Block = 8 waves sharing one m-tile (16 A rows): A tile async-staged into
// LDS once, each wave computes one 16x16 tile with a SW-pipelined K-loop.
// ---------------------------------------------------------------------------
template<int LAYER, int K>
__global__ __launch_bounds__(256, 2)
void gx_gemm(const __bf16* __restrict__ A0,
             const __bf16* __restrict__ A1,     // second K-half (layer1), else null
             const __bf16* __restrict__ W,      // [G x K]
             const float*  __restrict__ bias,   // [G]  (b_ih)
             float*        __restrict__ C)      // [Mtok x G]
{
    __shared__ __bf16 A_s[16 * K];              // 8KB (K=256) / 16KB (K=512)

    const int tid   = threadIdx.x;
    const int lane  = tid & 31;
    const int wave  = tid >> 5;
    const int nGrp  = (Gg / 16) / 8;            // 6 n-groups of 8 waves
    const int m_t   = blockIdx.x / nGrp;
    const int grp   = blockIdx.x - m_t * nGrp;
    const int n_t   = grp * 8 + wave;

    // ---- stage the 16 x K bf16 A tile into LDS (async, 16B chunks) -------
    const uint32_t a_s_base = (uint32_t)(uintptr_t)(&A_s[0]);
    constexpr int rowBytes = K * 2;
    constexpr int nChunk   = 16 * rowBytes / 16;     // 512 or 1024
    #pragma unroll
    for (int c = 0; c < nChunk / 256; ++c) {
        const int off    = (c * 256 + tid) * 16;     // byte offset in tile
        const int row    = off / rowBytes;
        const int within = off - row * rowBytes;
        const int tok    = m_t * 16 + row;
        const char* gsrc;
        if (LAYER == 0) {
            const int b = tok & (Bb - 1);
            const int s = tok / Bb;
            gsrc = (const char*)A0 + ((size_t)b * Ss + s) * (size_t)(Dd * 2) + within;
        } else {
            const char* base = (within < Hh * 2) ? (const char*)A0 : (const char*)A1;
            gsrc = base + (size_t)tok * (Hh * 2) + (within & (Hh * 2 - 1));
        }
        async_copy_b128(a_s_base + off, gsrc);
    }
    wait_async0();
    __syncthreads();

    const int half = lane >> 4;
    const int mn   = lane & 15;
    const int g    = n_t * 16 + mn;
    const __bf16* wrow = W + (size_t)g * K;
    const __bf16* arow = A_s + mn * K;

    v8f acc = {};
    const int ka0 = half * 8;
    v8bf alo = *(const v8bf*)(arow + ka0);
    v8bf ahi = *(const v8bf*)(arow + ka0 + 16);
    v8bf blo = *(const v8bf*)(wrow + ka0);
    v8bf bhi = *(const v8bf*)(wrow + ka0 + 16);
    #pragma unroll
    for (int k0 = 0; k0 < K; k0 += 32) {
        v8bf nalo, nahi, nblo, nbhi;
        if (k0 + 32 < K) {
            const int ka = k0 + 32 + half * 8;
            nalo = *(const v8bf*)(arow + ka);           // ds_load_b128
            nahi = *(const v8bf*)(arow + ka + 16);
            nblo = *(const v8bf*)(wrow + ka);           // L2-resident weights
            nbhi = *(const v8bf*)(wrow + ka + 16);
        }
        v16bf a = __builtin_shufflevector(alo, ahi, 0,1,2,3,4,5,6,7,8,9,10,11,12,13,14,15);
        v16bf b = __builtin_shufflevector(blo, bhi, 0,1,2,3,4,5,6,7,8,9,10,11,12,13,14,15);
        acc = __builtin_amdgcn_wmma_f32_16x16x32_bf16(false, a, false, b, (short)0, acc, false, false);
        alo = nalo; ahi = nahi; blo = nblo; bhi = nbhi;
    }

    const float bv = bias[g];
    #pragma unroll
    for (int r = 0; r < 8; ++r) {
        const int tok_out = m_t * 16 + half * 8 + r;    // C/D layout: m = r + 8*half
        C[(size_t)tok_out * Gg + g] = acc[r] + bv;
    }
}

// ---------------------------------------------------------------------------
// Persistent GRU scan. grid = {2} (dir 0 = fwd, 1 = bwd), 1024 threads.
// Tile map: wave w owns ALL THREE gate parts (r/z/n) of one 16x16 (b,j)
// patch: m_t = w&1, n_col = w>>1, B rows g = {0,256,512} + n_col*16.
// The gate math runs directly on the three register-resident accumulators --
// no gh LDS roundtrip at all. Per step:
//   wave 0 issues ONE tensor_load_to_lds (TDM) for the 96KB gx[t] slice
//   k-loop: 1 A-fragment (LDS) + 3 B-fragments (WGP$/L2) -> 3 WMMAs
//   s_wait_tensorcnt 0 (wave 0) ; barrier
//   fused gates from acc regs; h update; outputs; barrier
// LDS: h 2x16KB + gx 96KB + b_hh 3KB = 134KB of 320KB.
// The W_hh base is laundered through an empty asm as a scalar and re-cast to
// an addrspace(1) pointer: B loads stay inside the step loop (no LICM ->
// no 192-VGPR weight spill) AND lower to global_load_b128 (LOADcnt only),
// not flat_load (which would also tie up DScnt).
// ---------------------------------------------------------------------------
__global__ __launch_bounds__(1024, 1)
void gru_scan(const float*  __restrict__ gx,      // [S*B, G], tok = t*B+b
              const __bf16* __restrict__ Whh,     // [G x H] bf16
              const float*  __restrict__ bhh,     // [G]
              __bf16*       __restrict__ out_f,   // layer0 fwd states (bf16) or null
              __bf16*       __restrict__ out_b,   // layer0 bwd states (bf16) or null
              float*        __restrict__ out32,   // layer1: d_out [B,S,2H] or null
              float*        __restrict__ hn,      // d_out + B*S*2H : [4,B,H]
              int layer)
{
    extern __shared__ char smem[];
    __bf16* h_s   = (__bf16*)smem;                                   // [2][B*H]
    float*  gx_s  = (float*)(smem + 2 * Bb * Hh * sizeof(__bf16));   // [B*G]
    float*  bhh_s = gx_s + Bb * Gg;                                  // [G]

    const int tid   = threadIdx.x;
    const int dir   = blockIdx.x;
    const int lane  = tid & 31;
    const int wave  = tid >> 5;
    const int half  = lane >> 4;
    const int mn    = lane & 15;

    const int m_t   = wave & 1;          // which 16-row half of the batch
    const int n_col = wave >> 1;         // 0..15 -> columns j = n_col*16 + mn
    const int j     = n_col * 16 + mn;

    for (int i = tid; i < 2 * Bb * Hh; i += blockDim.x) h_s[i] = f2bf(0.0f);
    for (int i = tid; i < Gg; i += blockDim.x) bhh_s[i] = bhh[i];
    __syncthreads();

    const uint32_t gx_s_base = (uint32_t)(uintptr_t)gx_s;
    const float br = bhh_s[j];
    const float bz = bhh_s[j + Hh];
    const float bg = bhh_s[j + 2 * Hh];

    int cur = 0;
    for (int step = 0; step < Ss; ++step) {
        const int t = dir ? (Ss - 1 - step) : step;
        const __bf16* hc = h_s + cur * Bb * Hh;

        // ---- TDM: gx[t] slice (96KB = 12288 x 8B) -> LDS, one descriptor -
        if (wave == 0) {
            const uint64_t gsrc = (uint64_t)(uintptr_t)gx
                                + ((uint64_t)t * Bb) * Gg * sizeof(float);
            tdm_load_1d(gx_s_base, gsrc, (uint32_t)(Bb * Gg / 2));
        }

        // Launder the Whh base (scalar) so B loads stay inside the step loop.
        uint64_t wq = (uint64_t)(uintptr_t)Whh;
        asm volatile("" : "+s"(wq));
        const uint64_t wr0 = wq + (uint64_t)(0 * Hh + j) * (Hh * 2);  // r rows
        const uint64_t wr1 = wq + (uint64_t)(1 * Hh + j) * (Hh * 2);  // z rows
        const uint64_t wr2 = wq + (uint64_t)(2 * Hh + j) * (Hh * 2);  // n rows

        const __bf16* ar = hc + (m_t * 16 + mn) * Hh;

        // ---- GEMM: 3 gate parts of one 16x16 patch (overlaps the TDM) ---
        v8f acc0 = {}, acc1 = {}, acc2 = {};
        #pragma unroll
        for (int k0 = 0; k0 < Hh; k0 += 32) {
            const int ka  = k0 + half * 8;
            const int kb  = ka * 2;                        // byte offset
            v8bf alo = *(const v8bf*)(ar + ka);            // ds_load_b128
            v8bf ahi = *(const v8bf*)(ar + ka + 16);
            v16bf a = __builtin_shufflevector(alo, ahi, 0,1,2,3,4,5,6,7,8,9,10,11,12,13,14,15);
            v8bf b0lo = *(const gv8bf*)(uintptr_t)(wr0 + kb);        // global_load_b128
            v8bf b0hi = *(const gv8bf*)(uintptr_t)(wr0 + kb + 32);
            v16bf b0 = __builtin_shufflevector(b0lo, b0hi, 0,1,2,3,4,5,6,7,8,9,10,11,12,13,14,15);
            acc0 = __builtin_amdgcn_wmma_f32_16x16x32_bf16(false, a, false, b0, (short)0, acc0, false, false);
            v8bf b1lo = *(const gv8bf*)(uintptr_t)(wr1 + kb);
            v8bf b1hi = *(const gv8bf*)(uintptr_t)(wr1 + kb + 32);
            v16bf b1 = __builtin_shufflevector(b1lo, b1hi, 0,1,2,3,4,5,6,7,8,9,10,11,12,13,14,15);
            acc1 = __builtin_amdgcn_wmma_f32_16x16x32_bf16(false, a, false, b1, (short)0, acc1, false, false);
            v8bf b2lo = *(const gv8bf*)(uintptr_t)(wr2 + kb);
            v8bf b2hi = *(const gv8bf*)(uintptr_t)(wr2 + kb + 32);
            v16bf b2 = __builtin_shufflevector(b2lo, b2hi, 0,1,2,3,4,5,6,7,8,9,10,11,12,13,14,15);
            acc2 = __builtin_amdgcn_wmma_f32_16x16x32_bf16(false, a, false, b2, (short)0, acc2, false, false);
        }

        if (wave == 0) __builtin_amdgcn_s_wait_tensorcnt(0);  // gx[t] landed
        __syncthreads();                                      // publish gx_s

        // ---- fused gates straight from the accumulators ------------------
        #pragma unroll
        for (int r = 0; r < 8; ++r) {
            const int b = m_t * 16 + half * 8 + r;    // C/D layout: m = r + 8*half
            const float xr = gx_s[b * Gg + j];
            const float xz = gx_s[b * Gg + j + Hh];
            const float xn = gx_s[b * Gg + j + 2 * Hh];
            const float rr = 1.0f / (1.0f + __expf(-(xr + acc0[r] + br)));
            const float zz = 1.0f / (1.0f + __expf(-(xz + acc1[r] + bz)));
            const float nn = tanhf(xn + rr * (acc2[r] + bg));
            const float hp = bf2f(hc[b * Hh + j]);
            const float hv = (1.0f - zz) * nn + zz * hp;

            h_s[(cur ^ 1) * Bb * Hh + b * Hh + j] = f2bf(hv);
            if (layer == 0) {
                __bf16* ob = dir ? out_b : out_f;
                ob[((size_t)t * Bb + b) * Hh + j] = f2bf(hv);
            } else {
                out32[((size_t)b * Ss + t) * (2 * Hh) + dir * Hh + j] = hv;
            }
            if (step == Ss - 1)
                hn[((size_t)(layer * 2 + dir) * Bb + b) * Hh + j] = hv;
        }
        __syncthreads();
        cur ^= 1;
    }
}

// ---------------------------------------------------------------------------
extern "C" void kernel_launch(void* const* d_in, const int* in_sizes, int n_in,
                              void* d_out, int out_size, void* d_ws, size_t ws_size,
                              hipStream_t stream) {
    const float* x     = (const float*)d_in[0];
    const float* Wih0  = (const float*)d_in[1];
    const float* Whh0  = (const float*)d_in[2];
    const float* bih0  = (const float*)d_in[3];
    const float* bhh0  = (const float*)d_in[4];
    const float* Wih1  = (const float*)d_in[5];
    const float* Whh1  = (const float*)d_in[6];
    const float* bih1  = (const float*)d_in[7];
    const float* bhh1  = (const float*)d_in[8];

    // Workspace carve-up (~300 MB)
    char* ws = (char*)d_ws;
    size_t off = 0;
    auto carve = [&](size_t bytes) {
        void* p = ws + off;
        off += (bytes + 255) & ~(size_t)255;
        return p;
    };
    __bf16* xbf    = (__bf16*)carve((size_t)Bb * Ss * Dd * 2);
    __bf16* Wih0bf = (__bf16*)carve((size_t)Gg * Dd * 2);
    __bf16* Whh0bf = (__bf16*)carve((size_t)Gg * Hh * 2);
    __bf16* Wih1bf = (__bf16*)carve((size_t)Gg * 2 * Hh * 2);
    __bf16* Whh1bf = (__bf16*)carve((size_t)Gg * Hh * 2);
    __bf16* hsf    = (__bf16*)carve((size_t)Ss * Bb * Hh * 2);
    __bf16* hsb    = (__bf16*)carve((size_t)Ss * Bb * Hh * 2);
    float*  gxbuf  = (float*)carve((size_t)Ss * Bb * Gg * 4);

    float* out32 = (float*)d_out;
    float* hn    = out32 + (size_t)Bb * Ss * 2 * Hh;

    // 1) one-time fp32 -> bf16 conversions
    cvt_f32_bf16<<<4096, 256, 0, stream>>>(x,    xbf,    (size_t)Bb * Ss * Dd);
    cvt_f32_bf16<<<768,  256, 0, stream>>>(Wih0, Wih0bf, (size_t)Gg * Dd);
    cvt_f32_bf16<<<768,  256, 0, stream>>>(Whh0, Whh0bf, (size_t)Gg * Hh);
    cvt_f32_bf16<<<1536, 256, 0, stream>>>(Wih1, Wih1bf, (size_t)Gg * 2 * Hh);
    cvt_f32_bf16<<<768,  256, 0, stream>>>(Whh1, Whh1bf, (size_t)Gg * Hh);

    const int blocks = (Bb * Ss / 16) * ((Gg / 16) / 8);   // 4096 m-tiles * 6 n-groups
    const size_t smem = 2 * Bb * Hh * sizeof(__bf16)       // h double buffer
                      + (size_t)Bb * Gg * sizeof(float)    // gx staging
                      + Gg * sizeof(float);                // b_hh   = 134144 B

    // 2) layer 0: input projection + bidirectional scan
    gx_gemm<0, Dd><<<blocks, 256, 0, stream>>>(xbf, (const __bf16*)0, Wih0bf, bih0, gxbuf);
    gru_scan<<<2, 1024, smem, stream>>>(gxbuf, Whh0bf, bhh0, hsf, hsb, (float*)0, hn, 0);

    // 3) layer 1: projection of concat(hsf,hsb) + scan writing d_out
    gx_gemm<1, 2 * Hh><<<blocks, 256, 0, stream>>>(hsf, hsb, Wih1bf, bih1, gxbuf);
    gru_scan<<<2, 1024, smem, stream>>>(gxbuf, Whh1bf, bhh1, (__bf16*)0, (__bf16*)0, out32, hn, 1);
}
